// LSQuantLinearLayer_17532056502601
// MI455X (gfx1250) — 8-bit matrix operand order repaired and checked by reading, compiled, not run on hardware
//
#include <hip/hip_runtime.h>

typedef __attribute__((ext_vector_type(8))) int   v8i;
typedef __attribute__((ext_vector_type(4))) int   v4i;
typedef __attribute__((ext_vector_type(2))) int   v2i;

#define QL 127.0f
#define BM 128
#define BN 128
#define BK 64
#define LDSB 80

#if defined(__AMDGCN__) && __has_builtin(__builtin_amdgcn_global_load_async_to_lds_b128)
#define USE_ASYNC_LDS 1
typedef __attribute__((address_space(1))) v4i gv4i;
typedef __attribute__((address_space(3))) v4i lv4i;
#else
#define USE_ASYNC_LDS 0
#endif

__device__ __forceinline__ void wait_async0() {
#if __has_builtin(__builtin_amdgcn_s_wait_asynccnt)
  __builtin_amdgcn_s_wait_asynccnt(0);
#elif defined(__AMDGCN__)
  asm volatile("s_wait_asynccnt 0x0" ::: "memory");
#endif
}

__global__ void fq_pack_kernel(const float* __restrict__ x,
                               unsigned int* __restrict__ q,
                               const float* __restrict__ clip_max,
                               int cm_idx, long n4) {
  long i = (long)blockIdx.x * blockDim.x + threadIdx.x;
  if (i >= n4) return;
  const float cmax = clip_max[cm_idx];
  const float inv_scale = QL / cmax;
  const float4 v = ((const float4*)x)[i];
  const int q0 = (int)rintf(fminf(fmaxf(v.x, -cmax), cmax) * inv_scale);
  const int q1 = (int)rintf(fminf(fmaxf(v.y, -cmax), cmax) * inv_scale);
  const int q2 = (int)rintf(fminf(fmaxf(v.z, -cmax), cmax) * inv_scale);
  const int q3 = (int)rintf(fminf(fmaxf(v.w, -cmax), cmax) * inv_scale);
  q[i] = ((unsigned)q0 & 0xffu) | (((unsigned)q1 & 0xffu) << 8) |
         (((unsigned)q2 & 0xffu) << 16) | (((unsigned)q3 & 0xffu) << 24);
}

__global__ void __launch_bounds__(256, 2)
gemm_iu8(const char* __restrict__ qx, const char* __restrict__ qw,
         const float* __restrict__ bias, const float* __restrict__ clip_max,
         float* __restrict__ out, int M, int N, int K) {
  __shared__ char ldsA[2][BM * LDSB];
  __shared__ char ldsB[2][BN * LDSB];

  const int tid  = threadIdx.x;
  const int lane = tid & 31;
  const int wave = tid >> 5;
  const int wm   = wave >> 1;
  const int wn   = wave & 1;
  const int half = lane >> 4;
  const int l15  = lane & 15;

  const long block_m = (long)blockIdx.y * BM;
  const long block_n = (long)blockIdx.x * BN;

  const int ld_row = tid >> 2;
  const int ld_col = (tid & 3) << 4;

  const char* gA = qx + block_m * K;
  const char* gB = qw + block_n * K;

  int goff[2], soff[2];
#pragma unroll
  for (int h = 0; h < 2; ++h) {
    const int r = ld_row + h * 64;
    goff[h] = r * K + ld_col;
    soff[h] = r * LDSB + ld_col;
  }

  const char* aBase[2];
  const char* bBase[2];
#pragma unroll
  for (int b = 0; b < 2; ++b) {
    aBase[b] = &ldsA[b][(wm * 32 + l15) * LDSB + half * 16];
    bBase[b] = &ldsB[b][(wn * 64 + l15) * LDSB + half * 16];
  }

  v8i acc[2][4];
#pragma unroll
  for (int mi = 0; mi < 2; ++mi)
#pragma unroll
    for (int ni = 0; ni < 4; ++ni)
      acc[mi][ni] = 0;

  const int nk = K / BK;

#if USE_ASYNC_LDS
  auto stage = [&](int dstbuf, int k0) {
    const char* gAk = gA + k0;
    const char* gBk = gB + k0;
#pragma unroll
    for (int h = 0; h < 2; ++h) {
      __builtin_amdgcn_global_load_async_to_lds_b128(
          (gv4i*)(gAk + goff[h]), (lv4i*)&ldsA[dstbuf][soff[h]], 0, 0);
      __builtin_amdgcn_global_load_async_to_lds_b128(
          (gv4i*)(gBk + goff[h]), (lv4i*)&ldsB[dstbuf][soff[h]], 0, 0);
    }
  };
  stage(0, 0);
  wait_async0();
  __syncthreads();
#else
#pragma unroll
  for (int h = 0; h < 2; ++h) {
    *(v4i*)&ldsA[0][soff[h]] = *(const v4i*)(gA + goff[h]);
    *(v4i*)&ldsB[0][soff[h]] = *(const v4i*)(gB + goff[h]);
  }
  __syncthreads();
#endif

  auto step = [&](int buf, int kk, bool pf) {
#if USE_ASYNC_LDS
    if (pf) stage(buf ^ 1, (kk + 1) * BK);
#else
    v4i pa[2], pb[2];
    if (pf) {
      const char* gAk = gA + (kk + 1) * BK;
      const char* gBk = gB + (kk + 1) * BK;
#pragma unroll
      for (int h = 0; h < 2; ++h) {
        pa[h] = *(const v4i*)(gAk + goff[h]);
        pb[h] = *(const v4i*)(gBk + goff[h]);
      }
    }
#endif

    v8i a[2], b4[4];
#pragma unroll
    for (int mi = 0; mi < 2; ++mi) {
      const char* p = aBase[buf] + mi * (16 * LDSB);
      const v4i p0 = *(const v4i*)(p + 0);
      const v4i p1 = *(const v4i*)(p + 32);
      v8i f;
      f[0] = p0[0]; f[1] = p0[1]; f[2] = p0[2]; f[3] = p0[3];
      f[4] = p1[0]; f[5] = p1[1]; f[6] = p1[2]; f[7] = p1[3];
      a[mi] = f;
    }
#pragma unroll
    for (int ni = 0; ni < 4; ++ni) {
      const char* p = bBase[buf] + ni * (16 * LDSB);
      const v4i lo = *(const v4i*)(p + 0);
      const v4i hi = *(const v4i*)(p + 32);
      v8i f;
      f[0] = lo[0]; f[1] = lo[1]; f[2] = lo[2]; f[3] = lo[3];
      f[4] = hi[0]; f[5] = hi[1]; f[6] = hi[2]; f[7] = hi[3];
      b4[ni] = f;
    }

#pragma unroll
    for (int ni = 0; ni < 4; ++ni) {
#pragma unroll
      for (int mi = 0; mi < 2; ++mi) {
        acc[mi][ni] = __builtin_amdgcn_wmma_i32_16x16x64_iu8(
            true, a[mi], true, b4[ni], acc[mi][ni], false, false);
      }
    }

#if USE_ASYNC_LDS
    if (pf) wait_async0();
#else
    if (pf) {
#pragma unroll
      for (int h = 0; h < 2; ++h) {
        *(v4i*)&ldsA[buf ^ 1][soff[h]] = pa[h];
        *(v4i*)&ldsB[buf ^ 1][soff[h]] = pb[h];
      }
    }
#endif
    __syncthreads();
  };

  int kk = 0;
  for (; kk + 2 < nk; kk += 2) {
    step(0, kk, true);
    step(1, kk + 1, true);
  }
  if (kk + 1 < nk) {
    step(0, kk, true);
    step(1, kk + 1, false);
  } else {
    step(0, kk, false);
  }

  const float sc = (clip_max[0] / QL) * (clip_max[1] / QL);
  const unsigned uN = (unsigned)N;
#pragma unroll
  for (int ni = 0; ni < 4; ++ni) {
    const unsigned gn = (unsigned)block_n + wn * 64 + ni * 16 + l15;
    const float bv = bias[gn];
#pragma unroll
    for (int mi = 0; mi < 2; ++mi) {
      const unsigned gm0 = (unsigned)block_m + wm * 32 + mi * 16 + half * 8;
      float* po = out + (size_t)(gm0 * uN + gn);
#pragma unroll
      for (int r = 0; r < 8; ++r) {
        po[(size_t)((unsigned)r * uN)] = (float)acc[mi][ni][r] * sc + bv;
      }
    }
  }
}

extern "C" void kernel_launch(void* const* d_in, const int* in_sizes, int n_in,
                              void* d_out, int out_size, void* d_ws, size_t ws_size,
                              hipStream_t stream) {
  const float* x    = (const float*)d_in[0];
  const float* w    = (const float*)d_in[1];
  const float* bias = (const float*)d_in[2];
  const float* clip = (const float*)d_in[3];
  float* out = (float*)d_out;

  const int N = in_sizes[2];
  const int K = in_sizes[1] / N;
  const int M = (int)((long)in_sizes[0] / K);

  char* qx = (char*)d_ws;
  char* qw = qx + (size_t)M * K;

  const long nx4 = (long)M * K / 4;
  const long nw4 = (long)N * K / 4;
  fq_pack_kernel<<<(nx4 + 255) / 256, 256, 0, stream>>>(
      x, (unsigned int*)qx, clip, 0, nx4);
  fq_pack_kernel<<<(nw4 + 255) / 256, 256, 0, stream>>>(
      w, (unsigned int*)qw, clip, 1, nw4);

  dim3 grid(N / BN, M / BM);
  gemm_iu8<<<grid, 256, 0, stream>>>(qx, qw, bias, clip, out, M, N, K);
}
